// LearnableCosineVectorEmbedding_55473797595873
// MI455X (gfx1250) — matrix operand order repaired and checked by reading, not compiled, not run on hardware
//
#include <hip/hip_runtime.h>
#include <hip/hip_bf16.h>

typedef __attribute__((ext_vector_type(16))) __bf16 v16bf;
typedef __attribute__((ext_vector_type(8)))  __bf16 v8bf;
typedef __attribute__((ext_vector_type(4)))  __bf16 v4bf;
typedef __attribute__((ext_vector_type(2)))  __bf16 v2bf;
typedef __attribute__((ext_vector_type(8)))  float  v8f;

#define N_PROJ   64
#define NUM_BINS 20
#define INP_DIM  1024
#define EMB_DIM  1024
#define FEAT     (N_PROJ * NUM_BINS)   // 1280
#define TOKENS   (8 * 4096)            // 32768
#define SIGMA2   0.01f
#define EPSN     1e-12f

static __device__ __forceinline__ v16bf cat8(v8bf lo, v8bf hi) {
    return __builtin_shufflevector(lo, hi, 0,1,2,3,4,5,6,7,8,9,10,11,12,13,14,15);
}

// Async 16-byte global -> LDS copy (CDNA5 ASYNCcnt path).
static __device__ __forceinline__ void async_ld16(const void* gptr, void* lptr) {
    unsigned lds = (unsigned)(uintptr_t)lptr;   // low 32 bits of flat LDS addr = LDS offset
    asm volatile("global_load_async_to_lds_b128 %0, %1, off"
                 :
                 : "v"(lds), "v"(gptr)
                 : "memory");
}

// ---------------------------------------------------------------------------
// K0a: normalize proj_weight rows -> bf16  (64 rows x 1024)
// ---------------------------------------------------------------------------
__global__ __launch_bounds__(256) void norm_proj_kernel(const float* __restrict__ w,
                                                        __bf16* __restrict__ wn) {
    __shared__ float red[256];
    const int row = blockIdx.x;
    const int t   = threadIdx.x;
    float4 v = *(const float4*)&w[(size_t)row * INP_DIM + t * 4];
    red[t] = v.x * v.x + v.y * v.y + v.z * v.z + v.w * v.w;
    __syncthreads();
    for (int s = 128; s > 0; s >>= 1) {
        if (t < s) red[t] += red[t + s];
        __syncthreads();
    }
    const float inv = 1.f / fmaxf(sqrtf(red[0]), EPSN);
    v4bf o;
    o[0] = (__bf16)(v.x * inv); o[1] = (__bf16)(v.y * inv);
    o[2] = (__bf16)(v.z * inv); o[3] = (__bf16)(v.w * inv);
    *(v4bf*)&wn[(size_t)row * INP_DIM + t * 4] = o;
}

// ---------------------------------------------------------------------------
// K0b: convert emb_weight f32 -> bf16  (1024 x 1280)
// ---------------------------------------------------------------------------
__global__ __launch_bounds__(256) void conv_emb_kernel(const float* __restrict__ e,
                                                       __bf16* __restrict__ eb) {
    const size_t idx = ((size_t)blockIdx.x * 256 + threadIdx.x) * 4;
    float4 v = *(const float4*)&e[idx];
    v4bf o;
    o[0] = (__bf16)v.x; o[1] = (__bf16)v.y; o[2] = (__bf16)v.z; o[3] = (__bf16)v.w;
    *(v4bf*)&eb[idx] = o;
}

// ---------------------------------------------------------------------------
// K1: p = (x . wn) * (1/||x||)   via WMMA bf16.  M=32768, N=64, K=1024.
// One workgroup = 32 tokens; 8 waves cover 2(M) x 4(N) 16x16 tiles.
// Row norms of x accumulated during LDS staging, applied to the f32 accum.
// ---------------------------------------------------------------------------
__global__ __launch_bounds__(256) void gemm1_kernel(const float* __restrict__ x,
                                                    const __bf16* __restrict__ wn,
                                                    float* __restrict__ p) {
    __shared__ __align__(16) __bf16 a_lds[32][40];
    __shared__ float rnorm[32];

    const int t    = threadIdx.x;
    const int m0   = blockIdx.x * 32;
    const int lane = t & 31;
    const int w    = t >> 5;
    const int mt   = w & 1;
    const int nt   = w >> 1;
    const int half = lane >> 4;
    const int nn   = lane & 15;
    const int sr   = t >> 3;
    const int sc   = (t & 7) * 4;

    v8f acc = {};
    float ss = 0.f;

    for (int kt = 0; kt < 32; ++kt) {
        float4 xv = *(const float4*)&x[(size_t)(m0 + sr) * INP_DIM + kt * 32 + sc];
        ss += xv.x * xv.x + xv.y * xv.y + xv.z * xv.z + xv.w * xv.w;
        a_lds[sr][sc + 0] = (__bf16)xv.x;
        a_lds[sr][sc + 1] = (__bf16)xv.y;
        a_lds[sr][sc + 2] = (__bf16)xv.z;
        a_lds[sr][sc + 3] = (__bf16)xv.w;
        __syncthreads();

        const int arow = mt * 16 + nn;
        v8bf alo = *(const v8bf*)&a_lds[arow][half * 16];
        v8bf ahi = *(const v8bf*)&a_lds[arow][half * 16 + 8];
        v16bf a = cat8(alo, ahi);

        const __bf16* bp = &wn[(size_t)(nt * 16 + nn) * INP_DIM + kt * 32 + half * 16];
        v8bf blo = *(const v8bf*)bp;
        v8bf bhi = *(const v8bf*)(bp + 8);
        v16bf b = cat8(blo, bhi);

        acc = __builtin_amdgcn_wmma_f32_16x16x32_bf16(false, a, false, b,
                                                      (short)0, acc, false, false);
        __syncthreads();
    }

    ss += __shfl_xor(ss, 1, 32);
    ss += __shfl_xor(ss, 2, 32);
    ss += __shfl_xor(ss, 4, 32);
    if ((t & 7) == 0) rnorm[sr] = 1.f / fmaxf(sqrtf(ss), EPSN);
    __syncthreads();

#pragma unroll
    for (int r = 0; r < 8; ++r) {
        const int ml = mt * 16 + half * 8 + r;
        p[(size_t)(m0 + ml) * N_PROJ + nt * 16 + nn] = acc[r] * rnorm[ml];
    }
}

// ---------------------------------------------------------------------------
// K2a: RBF + top-4 mask -> z (bf16, unscaled) + per-token 1/||z|| scale.
// ---------------------------------------------------------------------------
__global__ __launch_bounds__(256) void rbf_topk_kernel(const float* __restrict__ p,
                                                       const float* __restrict__ mean,
                                                       __bf16* __restrict__ z,
                                                       float* __restrict__ scale) {
    const int t    = threadIdx.x;
    const int tokg = blockIdx.x * 32 + (t >> 3);
    const int pj   = t & 7;

    float ss = 0.f;
#pragma unroll
    for (int j = 0; j < 8; ++j) {
        const int proj = pj + j * 8;
        const float pv = p[(size_t)tokg * N_PROJ + proj];
        float acts[NUM_BINS];
        float m1 = -1e30f, m2 = -1e30f, m3 = -1e30f, m4 = -1e30f;
#pragma unroll
        for (int b = 0; b < NUM_BINS; ++b) {
            const float d = pv - mean[proj * NUM_BINS + b];
            const float a = __expf(-0.5f * d * d / SIGMA2);
            acts[b] = a;
            const float r1 = fminf(a, m1);  m1 = fmaxf(a, m1);
            const float r2 = fminf(r1, m2); m2 = fmaxf(r1, m2);
            const float r3 = fminf(r2, m3); m3 = fmaxf(r2, m3);
            m4 = fmaxf(r3, m4);
        }
        const float thr = m4;   // 4th largest (ties kept, matches top_k semantics)
#pragma unroll
        for (int b = 0; b < NUM_BINS; b += 2) {
            const float v0 = acts[b]     < thr ? 0.f : acts[b];
            const float v1 = acts[b + 1] < thr ? 0.f : acts[b + 1];
            ss += v0 * v0 + v1 * v1;
            v2bf pr; pr[0] = (__bf16)v0; pr[1] = (__bf16)v1;
            *(v2bf*)&z[(size_t)tokg * FEAT + proj * NUM_BINS + b] = pr;
        }
    }
    ss += __shfl_xor(ss, 1, 32);
    ss += __shfl_xor(ss, 2, 32);
    ss += __shfl_xor(ss, 4, 32);
    if ((t & 7) == 0) scale[tokg] = 1.f / fmaxf(sqrtf(ss), EPSN);
}

// ---------------------------------------------------------------------------
// K2b: out = diag(scale) * (z @ E^T)  — tiled WMMA GEMM, M=32768 N=1024 K=1280.
// BM=128, BN=128, BK=32.  8 waves as 2(M)x4(N); each wave: 4x2 16x16 tiles.
// Both operand tiles staged to LDS with GLOBAL_LOAD_ASYNC_TO_LDS_B128,
// double-buffered on ASYNCcnt.
// ---------------------------------------------------------------------------
#define BM 128
#define BN 128
#define BK 32
#define LSTR (BK + 8)                 // 40 bf16 row stride (bank-conflict pad)
#define KSTEPS (FEAT / BK)            // 40

__global__ __launch_bounds__(256) void gemm2_kernel(const __bf16* __restrict__ z,
                                                    const __bf16* __restrict__ eb,
                                                    const float* __restrict__ scale,
                                                    float* __restrict__ out) {
    __shared__ __align__(16) __bf16 a_lds[2][BM * LSTR];
    __shared__ __align__(16) __bf16 b_lds[2][BN * LSTR];

    const int t  = threadIdx.x;
    const int N0 = blockIdx.x * BN;   // gridDim.x = 8
    const int M0 = blockIdx.y * BM;   // gridDim.y = 256

    const __bf16* gA = z  + (size_t)M0 * FEAT;
    const __bf16* gB = eb + (size_t)N0 * FEAT;

    // staging assignment: thread t copies rows (t>>2) and (t>>2)+64,
    // 16B chunk at column (t&3)*8 of the BKx? tile.
    const int srow = t >> 2;
    const int scol = (t & 3) * 8;

    // issue one double-buffer batch (4 async b128 per thread)
    auto issue = [&](int buf, int kt) {
        const int kc = kt * BK;
        async_ld16(gA + (size_t)srow * FEAT + kc + scol,        &a_lds[buf][srow * LSTR + scol]);
        async_ld16(gA + (size_t)(srow + 64) * FEAT + kc + scol, &a_lds[buf][(srow + 64) * LSTR + scol]);
        async_ld16(gB + (size_t)srow * FEAT + kc + scol,        &b_lds[buf][srow * LSTR + scol]);
        async_ld16(gB + (size_t)(srow + 64) * FEAT + kc + scol, &b_lds[buf][(srow + 64) * LSTR + scol]);
    };

    const int lane = t & 31;
    const int w    = t >> 5;
    const int wm   = w & 1;          // 0..1 -> 64 rows each
    const int wn   = w >> 1;         // 0..3 -> 32 cols each
    const int half = lane >> 4;
    const int nn   = lane & 15;

    v8f acc[4][2] = {};

    issue(0, 0);
    issue(1, 1);

#pragma unroll 1
    for (int kt = 0; kt < KSTEPS; ++kt) {
        const int cur = kt & 1;
        if (kt < KSTEPS - 1) {
            asm volatile("s_wait_asynccnt 0x4" ::: "memory");  // oldest batch retired
        } else {
            asm volatile("s_wait_asynccnt 0x0" ::: "memory");
        }
        __syncthreads();

        // B fragments: elem i -> k = half*16 + i
        v16bf bf0, bf1;
        {
            const __bf16* r0 = &b_lds[cur][(wn * 32 + nn) * LSTR + half * 16];
            bf0 = cat8(*(const v8bf*)r0, *(const v8bf*)(r0 + 8));
            const __bf16* r1 = &b_lds[cur][(wn * 32 + 16 + nn) * LSTR + half * 16];
            bf1 = cat8(*(const v8bf*)r1, *(const v8bf*)(r1 + 8));
        }
        // A fragments: elem i -> k = half*16 + i
#pragma unroll
        for (int s = 0; s < 4; ++s) {
            const __bf16* ra = &a_lds[cur][(wm * 64 + s * 16 + nn) * LSTR];
            v16bf a = cat8(*(const v8bf*)(ra + half * 16), *(const v8bf*)(ra + half * 16 + 8));
            acc[s][0] = __builtin_amdgcn_wmma_f32_16x16x32_bf16(false, a, false, bf0,
                                                                (short)0, acc[s][0], false, false);
            acc[s][1] = __builtin_amdgcn_wmma_f32_16x16x32_bf16(false, a, false, bf1,
                                                                (short)0, acc[s][1], false, false);
        }
        __syncthreads();
        if (kt + 2 < KSTEPS) issue(cur, kt + 2);
    }

#pragma unroll
    for (int s = 0; s < 4; ++s) {
#pragma unroll
        for (int r = 0; r < 8; ++r) {
            const int m = M0 + wm * 64 + s * 16 + half * 8 + r;
            const float sc = scale[m];
            out[(size_t)m * EMB_DIM + N0 + wn * 32 + nn]      = acc[s][0][r] * sc;
            out[(size_t)m * EMB_DIM + N0 + wn * 32 + 16 + nn] = acc[s][1][r] * sc;
        }
    }
}

// ---------------------------------------------------------------------------
// Launch
// ---------------------------------------------------------------------------
extern "C" void kernel_launch(void* const* d_in, const int* in_sizes, int n_in,
                              void* d_out, int out_size, void* d_ws, size_t ws_size,
                              hipStream_t stream) {
    const float* x    = (const float*)d_in[0];  // [8,4096,1024]
    const float* pw   = (const float*)d_in[1];  // [64,1024]
    const float* mean = (const float*)d_in[2];  // [1,1,64,20]
    const float* ew   = (const float*)d_in[3];  // [1024,1280]
    float* out        = (float*)d_out;          // [8,4096,1024]

    char* ws = (char*)d_ws;
    size_t off = 0;
    __bf16* wn = (__bf16*)(ws + off); off += (size_t)N_PROJ * INP_DIM * 2;   // 128 KB
    __bf16* eb = (__bf16*)(ws + off); off += (size_t)EMB_DIM * FEAT * 2;     // 2.5 MB
    float*  pp = (float*)(ws + off);  off += (size_t)TOKENS * N_PROJ * 4;    // 8 MB
    __bf16* zz = (__bf16*)(ws + off); off += (size_t)TOKENS * FEAT * 2;      // 80 MB
    float*  sc = (float*)(ws + off);  off += (size_t)TOKENS * 4;             // 128 KB

    norm_proj_kernel<<<N_PROJ, 256, 0, stream>>>(pw, wn);
    conv_emb_kernel<<<(EMB_DIM * FEAT) / 1024, 256, 0, stream>>>(ew, eb);
    gemm1_kernel<<<TOKENS / 32, 256, 0, stream>>>(x, wn, pp);
    rbf_topk_kernel<<<TOKENS / 32, 256, 0, stream>>>(pp, mean, zz, sc);
    gemm2_kernel<<<dim3(EMB_DIM / BN, TOKENS / BM), 256, 0, stream>>>(zz, eb, sc, out);
}
